// ShapeTokenizeEncoder_28991029248848
// MI455X (gfx1250) — matrix operand order repaired and checked by reading, not compiled, not run on hardware
//
#include <hip/hip_runtime.h>

// ---------------------------------------------------------------------------
// CDNA5 (gfx1250, wave32) fused PointNet++ encoder.
// GEMMs use v_wmma_f32_16x16x32_f16; activations live in LDS per group-block.
// B (weight) fragments are reused across MT M-tiles per wave to cut L2 traffic.
// ---------------------------------------------------------------------------

typedef __attribute__((ext_vector_type(8)))  _Float16 v8h;
typedef __attribute__((ext_vector_type(16))) _Float16 v16h;
typedef __attribute__((ext_vector_type(8)))  float    v8f;

union V16 { v16h v; v8h h[2]; };

// A fragment (16x32 f16, MxK): lanes 0-15 -> M=lane, K 0-7 & 16-23;
// lanes 16-31 -> M=lane-16, K 8-15 & 24-31.  Two aligned 128-bit loads.
__device__ __forceinline__ v16h ld_a16x32(const _Float16* act, int lane, int ldk, int k0) {
  int m = lane & 15;
  int g = lane >> 4;
  const _Float16* p = act + (size_t)m * ldk + (size_t)(k0 + g * 8);
  V16 u;
  u.h[0] = *(const v8h*)(p);        // k = k0+g*8   .. +7
  u.h[1] = *(const v8h*)(p + 16);   // k = k0+g*8+16.. +23
  return u.v;
}

// B fragment (32x16 f16, KxN), B[k][n] = W[n][k] (W row-major [cout][kpad]):
// lane n = lane&15, K 0-7 & 16-23 (lanes 0-15), K 8-15 & 24-31 (lanes 16-31).
__device__ __forceinline__ v16h ld_b32x16(const _Float16* W, int lane, int ldk, int nt, int k0) {
  int n = (lane & 15) + nt * 16;
  int g = lane >> 4;
  const _Float16* p = W + (size_t)n * ldk + (size_t)(k0 + g * 8);
  V16 u;
  u.h[0] = *(const v8h*)(p);
  u.h[1] = *(const v8h*)(p + 16);
  return u.v;
}

// One 1x1-conv + BN + ReLU layer; each wave owns MT consecutive 16-row tiles.
// B fragment loaded once per (nt,k0) and reused for MT WMMAs.
template<int KP, int CO, int MT>
__device__ __forceinline__ void mlp_layer_mt(const _Float16* actIn, _Float16* actOut,
                                             const _Float16* __restrict__ W,
                                             const float* __restrict__ sb,
                                             int lane, int wave)
{
  const _Float16* a0 = actIn + wave * MT * 16 * KP;
  _Float16* o0 = actOut + wave * MT * 16 * CO;
  const v8f vzero = {0.f, 0.f, 0.f, 0.f, 0.f, 0.f, 0.f, 0.f};
  #pragma unroll 1
  for (int nt = 0; nt < CO / 16; ++nt) {
    v8f acc[MT];
    #pragma unroll
    for (int m = 0; m < MT; ++m) acc[m] = vzero;
    #pragma unroll
    for (int k0 = 0; k0 < KP; k0 += 32) {
      v16h b = ld_b32x16(W, lane, KP, nt, k0);
      #pragma unroll
      for (int m = 0; m < MT; ++m) {
        v16h a = ld_a16x32(a0 + m * 16 * KP, lane, KP, k0);
        acc[m] = __builtin_amdgcn_wmma_f32_16x16x32_f16(false, a, false, b,
                                                        (short)0, acc[m], false, false);
      }
    }
    int n = (lane & 15) + nt * 16;
    float sc = sb[2 * n];
    float bi = sb[2 * n + 1];
    int mr = (lane >> 4) << 3;          // C/D layout: M = r + 8*(lane>=16)
    #pragma unroll
    for (int m = 0; m < MT; ++m) {
      #pragma unroll
      for (int r = 0; r < 8; ++r) {
        float v = acc[m][r] * sc + bi;
        o0[(m * 16 + mr + r) * CO + n] = (_Float16)(v > 0.f ? v : 0.f);
      }
    }
  }
}

// --------------------------- weight prep -----------------------------------
// fp32 [cout][cin] -> f16 [cout][kpad] (zero pad), plus fused BN scale/bias.
__global__ __launch_bounds__(256) void prep_kernel(
    const float* __restrict__ W, const float* __restrict__ gamma,
    const float* __restrict__ beta, const float* __restrict__ mean,
    const float* __restrict__ var, _Float16* __restrict__ Wh,
    float* __restrict__ sb, int cout, int cin, int kpad)
{
  int i = blockIdx.x * 256 + threadIdx.x;
  int total = cout * kpad;
  if (i < total) {
    int n = i / kpad, k = i - n * kpad;
    Wh[i] = (_Float16)(k < cin ? W[(size_t)n * cin + k] : 0.f);
  }
  if (sb != nullptr && i < cout) {
    float sc = gamma[i] * rsqrtf(var[i] + 1e-5f);
    sb[2 * i]     = sc;
    sb[2 * i + 1] = beta[i] - mean[i] * sc;
  }
}

// ------------------------------- FPS ---------------------------------------
// One block per batch; sequential farthest-point selection, LDS argmax.
__global__ __launch_bounds__(256) void fps_kernel(
    const float* __restrict__ xyz, int xstride, int N, int S,
    int* __restrict__ outIdx, float* __restrict__ outXyz)
{
  int b = blockIdx.x, tid = threadIdx.x;
  __shared__ float dist[1024];
  __shared__ float rmax[256];
  __shared__ int   rid[256];
  __shared__ float cent[3];
  __shared__ int   sfar;
  const float* base = xyz + (size_t)b * N * xstride;
  for (int j = tid; j < N; j += 256) dist[j] = 1e10f;
  if (tid == 0) sfar = 0;
  __syncthreads();
  for (int s = 0; s < S; ++s) {
    if (tid == 0) {
      int f = sfar;
      outIdx[(size_t)b * S + s] = f;
      float c0 = base[(size_t)f * xstride + 0];
      float c1 = base[(size_t)f * xstride + 1];
      float c2 = base[(size_t)f * xstride + 2];
      cent[0] = c0; cent[1] = c1; cent[2] = c2;
      float* o = outXyz + ((size_t)b * S + s) * 3;
      o[0] = c0; o[1] = c1; o[2] = c2;
    }
    __syncthreads();
    float bm = -1.f; int bi = 0;
    for (int j = tid; j < N; j += 256) {
      float dx = base[(size_t)j * xstride + 0] - cent[0];
      float dy = base[(size_t)j * xstride + 1] - cent[1];
      float dz = base[(size_t)j * xstride + 2] - cent[2];
      float d  = fminf(dist[j], dx * dx + dy * dy + dz * dz);
      dist[j] = d;
      if (d > bm) { bm = d; bi = j; }
    }
    rmax[tid] = bm; rid[tid] = bi;
    __syncthreads();
    for (int off = 128; off > 0; off >>= 1) {
      if (tid < off) {
        float o = rmax[tid + off]; int oi = rid[tid + off];
        if (o > rmax[tid] || (o == rmax[tid] && oi < rid[tid])) { rmax[tid] = o; rid[tid] = oi; }
      }
      __syncthreads();
    }
    if (tid == 0) sfar = rid[0];
    __syncthreads();
  }
}

// ---------------------------- ball query -----------------------------------
// One thread per (b,s): first 32 points (index order) with d2 < r2,
// padded with the first hit (matches pointnet2 CUDA semantics).
__global__ __launch_bounds__(256) void ballq_kernel(
    const float* __restrict__ xyz, int xstride, int N,
    const float* __restrict__ newxyz, int S, float r2,
    int* __restrict__ gidx, int BS)
{
  int t = blockIdx.x * 256 + threadIdx.x;
  if (t >= BS) return;
  int b = t / S;
  const float* base = xyz + (size_t)b * N * xstride;
  float cx = newxyz[(size_t)t * 3 + 0];
  float cy = newxyz[(size_t)t * 3 + 1];
  float cz = newxyz[(size_t)t * 3 + 2];
  int* o = gidx + (size_t)t * 32;
  int cnt = 0, first = 0;
  for (int j = 0; j < N && cnt < 32; ++j) {
    float dx = base[(size_t)j * xstride + 0] - cx;
    float dy = base[(size_t)j * xstride + 1] - cy;
    float dz = base[(size_t)j * xstride + 2] - cz;
    if (dx * dx + dy * dy + dz * dz < r2) {
      if (cnt == 0) first = j;
      o[cnt++] = j;
    }
  }
  for (; cnt < 32; ++cnt) o[cnt] = first;
}

// --------------------------- SA1 fused MLP ---------------------------------
// block = 4 groups (128 rows); 2 waves, each owns MT=4 tiles (64 rows).
__global__ __launch_bounds__(64) void sa1_mlp_kernel(
    const float* __restrict__ pc, const int* __restrict__ gidx,
    const float* __restrict__ newxyz,
    const _Float16* __restrict__ w0, const _Float16* __restrict__ w1,
    const _Float16* __restrict__ w2,
    const float* __restrict__ sb0, const float* __restrict__ sb1,
    const float* __restrict__ sb2, float* __restrict__ featsOut)
{
  constexpr int G = 4, R = G * 32, K0 = 32, C1 = 64, C2 = 64, C3 = 128, MT = 4;
  __shared__ alignas(16) _Float16 bufA[R * C2];  // K0 input (R*32), then C2 (R*64)
  __shared__ alignas(16) _Float16 bufB[R * C3];  // C1 (R*64), then C3 (R*128)
  int g0 = blockIdx.x * G;       // first group = b*32 + s
  int tid = threadIdx.x, lane = tid & 31, wave = tid >> 5;
  for (int t = tid; t < R * K0; t += 64) {
    int row = t / K0, k = t - row * K0;
    int g = g0 + (row >> 5);
    int b = g >> 5;
    int gi = gidx[(size_t)g * 32 + (row & 31)];
    float v = 0.f;
    if (k < 3)      v = pc[((size_t)b * 1024 + gi) * 6 + k] - newxyz[(size_t)g * 3 + k];
    else if (k < 6) v = pc[((size_t)b * 1024 + gi) * 6 + k];   // feats = pc[...,3:]
    bufA[t] = (_Float16)v;
  }
  __syncthreads();
  mlp_layer_mt<K0, C1, MT>(bufA, bufB, w0, sb0, lane, wave);
  __syncthreads();
  mlp_layer_mt<C1, C2, MT>(bufB, bufA, w1, sb1, lane, wave);
  __syncthreads();
  mlp_layer_mt<C2, C3, MT>(bufA, bufB, w2, sb2, lane, wave);
  __syncthreads();
  for (int t = tid; t < G * C3; t += 64) {
    int grp = t / C3, c = t - grp * C3;
    float m = 0.f;
    for (int n = 0; n < 32; ++n) m = fmaxf(m, (float)bufB[(grp * 32 + n) * C3 + c]);
    featsOut[(size_t)(g0 + grp) * C3 + c] = m;
  }
}

// --------------------------- SA2 fused MLP ---------------------------------
// block = 2 groups (64 rows); 2 waves, each owns MT=2 tiles (32 rows).
__global__ __launch_bounds__(64) void sa2_mlp_kernel(
    const float* __restrict__ xyz1, const float* __restrict__ feats1,
    const int* __restrict__ gidx, const float* __restrict__ newxyz2,
    const _Float16* __restrict__ w0, const _Float16* __restrict__ w1,
    const _Float16* __restrict__ w2,
    const float* __restrict__ sb0, const float* __restrict__ sb1,
    const float* __restrict__ sb2, float* __restrict__ featsOut)
{
  constexpr int G = 2, R = G * 32, K0 = 160, C1 = 128, C2 = 128, C3 = 256, MT = 2;
  __shared__ alignas(16) _Float16 bufA[R * K0];  // K0 input (R*160), then C2 (R*128)
  __shared__ alignas(16) _Float16 bufB[R * C3];  // C1 (R*128), then C3 (R*256)
  int g0 = blockIdx.x * G;       // first group = b*16 + s
  int tid = threadIdx.x, lane = tid & 31, wave = tid >> 5;
  for (int t = tid; t < R * K0; t += 64) {
    int row = t / K0, k = t - row * K0;
    int g = g0 + (row >> 5);
    int b = g >> 4;
    int gi = gidx[(size_t)g * 32 + (row & 31)];
    float v = 0.f;
    if (k < 3)        v = xyz1[((size_t)b * 32 + gi) * 3 + k] - newxyz2[(size_t)g * 3 + k];
    else if (k < 131) v = feats1[((size_t)b * 32 + gi) * 128 + (k - 3)];
    bufA[t] = (_Float16)v;
  }
  __syncthreads();
  mlp_layer_mt<K0, C1, MT>(bufA, bufB, w0, sb0, lane, wave);
  __syncthreads();
  mlp_layer_mt<C1, C2, MT>(bufB, bufA, w1, sb1, lane, wave);
  __syncthreads();
  mlp_layer_mt<C2, C3, MT>(bufA, bufB, w2, sb2, lane, wave);
  __syncthreads();
  for (int t = tid; t < G * C3; t += 64) {
    int grp = t / C3, c = t - grp * C3;
    float m = 0.f;
    for (int n = 0; n < 32; ++n) m = fmaxf(m, (float)bufB[(grp * 32 + n) * C3 + c]);
    featsOut[(size_t)(g0 + grp) * C3 + c] = m;
  }
}

// ------------------------- SA3 (global) fused MLP ---------------------------
// block = 4 batches (64 rows); 2 waves, each owns MT=2 tiles (2 batches).
__global__ __launch_bounds__(64) void sa3_mlp_kernel(
    const float* __restrict__ newxyz2, const float* __restrict__ feats2,
    const _Float16* __restrict__ w0, const _Float16* __restrict__ w1,
    const _Float16* __restrict__ w2,
    const float* __restrict__ sb0, const float* __restrict__ sb1,
    const float* __restrict__ sb2, _Float16* __restrict__ feats3h)
{
  constexpr int G = 4, R = G * 16, K0 = 288, C1 = 256, C2 = 512, C3 = 768, MT = 2;
  __shared__ alignas(16) _Float16 bufA[R * C2];  // K0 (R*288) then C2 (R*512) : 64 KB
  __shared__ alignas(16) _Float16 bufB[R * C3];  // C1 (R*256) then C3 (R*768) : 96 KB
  int b0 = blockIdx.x * G;
  int tid = threadIdx.x, lane = tid & 31, wave = tid >> 5;
  for (int t = tid; t < R * K0; t += 64) {
    int row = t / K0, k = t - row * K0;
    int b = b0 + (row >> 4), n = row & 15;
    float v = 0.f;
    if (k < 3)        v = newxyz2[((size_t)b * 16 + n) * 3 + k];   // absolute xyz
    else if (k < 259) v = feats2[((size_t)b * 16 + n) * 256 + (k - 3)];
    bufA[t] = (_Float16)v;
  }
  __syncthreads();
  mlp_layer_mt<K0, C1, MT>(bufA, bufB, w0, sb0, lane, wave);
  __syncthreads();
  mlp_layer_mt<C1, C2, MT>(bufB, bufA, w1, sb1, lane, wave);
  __syncthreads();
  mlp_layer_mt<C2, C3, MT>(bufA, bufB, w2, sb2, lane, wave);
  __syncthreads();
  for (int t = tid; t < G * C3; t += 64) {
    int grp = t / C3, c = t - grp * C3;
    float m = 0.f;
    for (int n = 0; n < 16; ++n) m = fmaxf(m, (float)bufB[(grp * 16 + n) * C3 + c]);
    feats3h[(size_t)(b0 + grp) * C3 + c] = (_Float16)m;
  }
}

// ------------------------------- FC GEMM -----------------------------------
// out[1024,768] = X[1024,768] @ W^T + b.
// Each wave computes a 64x16 strip (MT=4 tiles sharing each B fragment).
__global__ __launch_bounds__(128) void fc_kernel(
    const _Float16* __restrict__ X, const _Float16* __restrict__ W,
    const float* __restrict__ bias, float* __restrict__ out)
{
  constexpr int K = 768, NT = 48, MT = 4;  // 1024/64 x 768/16 = 16x48 strips
  int tid = threadIdx.x, lane = tid & 31, wave = tid >> 5;
  int t = blockIdx.x * 4 + wave;           // 768 strips total
  int mt64 = t / NT, nt = t - mt64 * NT;
  const _Float16* a0 = X + (size_t)mt64 * 64 * K;
  const v8f vzero = {0.f, 0.f, 0.f, 0.f, 0.f, 0.f, 0.f, 0.f};
  v8f acc[MT];
  #pragma unroll
  for (int m = 0; m < MT; ++m) acc[m] = vzero;
  #pragma unroll 2
  for (int k0 = 0; k0 < K; k0 += 32) {
    v16h b = ld_b32x16(W, lane, K, nt, k0);
    #pragma unroll
    for (int m = 0; m < MT; ++m) {
      v16h a = ld_a16x32(a0 + m * 16 * K, lane, K, k0);
      acc[m] = __builtin_amdgcn_wmma_f32_16x16x32_f16(false, a, false, b,
                                                      (short)0, acc[m], false, false);
    }
  }
  int n = (lane & 15) + nt * 16;
  float bi = bias[n];
  #pragma unroll
  for (int m = 0; m < MT; ++m) {
    int m0 = mt64 * 64 + m * 16 + ((lane >> 4) << 3);
    #pragma unroll
    for (int r = 0; r < 8; ++r)
      out[(size_t)(m0 + r) * 768 + n] = acc[m][r] + bi;
  }
}

// ------------------------------ launcher -----------------------------------
extern "C" void kernel_launch(void* const* d_in, const int* in_sizes, int n_in,
                              void* d_out, int out_size, void* d_ws, size_t ws_size,
                              hipStream_t stream) {
  (void)in_sizes; (void)n_in; (void)out_size; (void)ws_size;
  const float* pc = (const float*)d_in[0];

  // layer dims: cout, cin, kpad (kpad = cin padded to multiple of 32)
  const int co[10] = {64, 64, 128, 128, 128, 256, 256, 512, 768, 768};
  const int ci[10] = {6, 64, 64, 131, 128, 128, 259, 256, 512, 768};
  const int kp[10] = {32, 64, 64, 160, 128, 128, 288, 256, 512, 768};

  char* ws = (char*)d_ws;
  size_t off = 0;
  auto take = [&](size_t bytes) -> char* {
    char* p = ws + off;
    off = (off + bytes + 255) & ~(size_t)255;
    return p;
  };
  _Float16* wh[10];
  for (int l = 0; l < 10; ++l) wh[l] = (_Float16*)take((size_t)co[l] * kp[l] * 2);
  float* sb[9];
  for (int l = 0; l < 9; ++l) sb[l] = (float*)take((size_t)co[l] * 2 * 4);
  int*      fid1 = (int*)take((size_t)1024 * 32 * 4);
  float*    nx1  = (float*)take((size_t)1024 * 32 * 3 * 4);
  int*      gi1  = (int*)take((size_t)1024 * 32 * 32 * 4);
  float*    f1   = (float*)take((size_t)1024 * 32 * 128 * 4);
  int*      fid2 = (int*)take((size_t)1024 * 16 * 4);
  float*    nx2  = (float*)take((size_t)1024 * 16 * 3 * 4);
  int*      gi2  = (int*)take((size_t)1024 * 16 * 32 * 4);
  float*    f2   = (float*)take((size_t)1024 * 16 * 256 * 4);
  _Float16* f3h  = (_Float16*)take((size_t)1024 * 768 * 2);

  // d_in layout (setup_inputs insertion order):
  // [0]=pc, then 9 layers x {W, gamma, beta, mean, var}, [46]=fc_w, [47]=fc_b.
  for (int l = 0; l < 9; ++l) {
    int base = 1 + l * 5;
    int total = co[l] * kp[l];
    prep_kernel<<<(total + 255) / 256, 256, 0, stream>>>(
        (const float*)d_in[base + 0], (const float*)d_in[base + 1],
        (const float*)d_in[base + 2], (const float*)d_in[base + 3],
        (const float*)d_in[base + 4], wh[l], sb[l], co[l], ci[l], kp[l]);
  }
  prep_kernel<<<(768 * 768 + 255) / 256, 256, 0, stream>>>(
      (const float*)d_in[46], nullptr, nullptr, nullptr, nullptr,
      wh[9], nullptr, 768, 768, 768);
  const float* fcb = (const float*)d_in[47];

  // ---- SA1 ----
  fps_kernel<<<1024, 256, 0, stream>>>(pc, 6, 1024, 32, fid1, nx1);
  ballq_kernel<<<(32768 + 255) / 256, 256, 0, stream>>>(
      pc, 6, 1024, nx1, 32, 0.04f, gi1, 32768);
  sa1_mlp_kernel<<<32768 / 4, 64, 0, stream>>>(
      pc, gi1, nx1, wh[0], wh[1], wh[2], sb[0], sb[1], sb[2], f1);

  // ---- SA2 ----
  fps_kernel<<<1024, 256, 0, stream>>>(nx1, 3, 32, 16, fid2, nx2);
  ballq_kernel<<<(16384 + 255) / 256, 256, 0, stream>>>(
      nx1, 3, 32, nx2, 16, 0.16f, gi2, 16384);
  sa2_mlp_kernel<<<16384 / 2, 64, 0, stream>>>(
      nx1, f1, gi2, nx2, wh[3], wh[4], wh[5], sb[3], sb[4], sb[5], f2);

  // ---- SA3 (global) ----
  sa3_mlp_kernel<<<1024 / 4, 64, 0, stream>>>(
      nx2, f2, wh[6], wh[7], wh[8], sb[6], sb[7], sb[8], f3h);

  // ---- FC ----
  fc_kernel<<<768 / 4, 128, 0, stream>>>(f3h, wh[9], fcb, (float*)d_out);
}
